// GlobalPointerLayer_60112362275159
// MI455X (gfx1250) — compile-verified, matrix operand order checked by reading, not run on hardware
//
#include <hip/hip_runtime.h>
#include <hip/hip_bf16.h>

typedef __attribute__((ext_vector_type(16))) _Float16 v16h;
typedef __attribute__((ext_vector_type(8)))  float    v8f;

#define HEADS 12
#define HSZ   64
#define LSEQ  512
#define DIM   768
#define NPROJ 1536
#define BATCH 32
#define NEGC  1000000000000.0f

// Pack two f32 into packed f16 (v_cvt_pk_rtz_f16_f32).
__device__ __forceinline__ unsigned pk2(float a, float b) {
#if __has_builtin(__builtin_amdgcn_cvt_pkrtz)
    auto r = __builtin_amdgcn_cvt_pkrtz(a, b);
    return __builtin_bit_cast(unsigned, r);
#else
    union { _Float16 h[2]; unsigned u; } x;
    x.h[0] = (_Float16)a; x.h[1] = (_Float16)b;
    return x.u;
#endif
}

// Load a 16-half WMMA fragment (A-layout) from LDS.
// rowp = this lane's 32-half K-row (row M=lane&15 for A, col N=lane&15 for B).
// Lanes 0-15 take K = {0..7, 16..23}; lanes 16-31 take K = {8..15, 24..31}.
__device__ __forceinline__ v16h frag16(const _Float16* rowp, int lane) {
    const int kb = (lane >> 4) << 3;   // 0 or 8 halves
    union { uint4 q[2]; v16h h; } u;
    u.q[0] = *(const uint4*)(rowp + kb);        // VGPR0-3: K = kb..kb+7
    u.q[1] = *(const uint4*)(rowp + 16 + kb);   // VGPR4-7: K = 16+kb..16+kb+7
    return u.h;
}

// ---------------------------------------------------------------------------
// Kernel 0: RoPE sin/cos table, tab[l][i] = (sin(l*freq_i), cos(l*freq_i)).
// ---------------------------------------------------------------------------
__global__ void rope_table_kernel(float2* __restrict__ tab) {
    const int l = blockIdx.x;        // 0..511
    const int i = threadIdx.x;       // 0..31
    const float freq = __expf(-9.2103403719761836f * (float)(2 * i) * (1.0f / 64.0f));
    const float ang  = (float)l * freq;
    tab[l * 32 + i] = make_float2(sinf(ang), cosf(ang));
}

// ---------------------------------------------------------------------------
// Kernel 1: proj = x@W + b, RoPE, pack q/k as f16 into workspace.
// Block = 256 threads (8 waves). Tile: 128 rows (M) x 128 cols (N), K-step 32.
// Wave = 32M x 64N (acc[2][4]); B-fragments reused across the two M subtiles.
// ---------------------------------------------------------------------------
__global__ __launch_bounds__(256) void proj_rope_kernel(
    const float* __restrict__ x, const float* __restrict__ W,
    const float* __restrict__ bias, const float2* __restrict__ tab,
    _Float16* __restrict__ qb, _Float16* __restrict__ kb) {
    __shared__ __align__(16) _Float16 Xs[128 * 32];   // [m][k]
    __shared__ __align__(16) _Float16 Ws[128 * 32];   // [n][k] (transposed)

    const int tid  = threadIdx.x;
    const int lane = tid & 31;
    const int w    = tid >> 5;
    const int mBase = blockIdx.x * 128;
    const int nBase = blockIdx.y * 128;
    const int wm = (w & 3) * 32;    // wave's M offset (2 subtiles)
    const int wn = (w >> 2) * 64;   // wave's N offset (4 subtiles)

    v8f acc[2][4] = {};

    for (int ks = 0; ks < DIM; ks += 32) {
        // Stage x tile 128x32: each thread packs 16 consecutive f32 -> 2x b128.
        {
            const int r  = tid >> 1;           // 0..127
            const int cb = (tid & 1) * 16;     // 0 or 16
            const float4* src = (const float4*)(x + (size_t)(mBase + r) * DIM + ks + cb);
            const float4 f0 = src[0], f1 = src[1], f2 = src[2], f3 = src[3];
            uint4 u0, u1;
            u0.x = pk2(f0.x, f0.y); u0.y = pk2(f0.z, f0.w);
            u0.z = pk2(f1.x, f1.y); u0.w = pk2(f1.z, f1.w);
            u1.x = pk2(f2.x, f2.y); u1.y = pk2(f2.z, f2.w);
            u1.z = pk2(f3.x, f3.y); u1.w = pk2(f3.z, f3.w);
            uint4* dst = (uint4*)(Xs + r * 32 + cb);
            dst[0] = u0; dst[1] = u1;
        }
        // Stage W tile 32x128 transposed: thread owns row n, 16 consecutive k.
        // Global reads column-strided per thread but coalesced across the 128
        // adjacent-n lanes; LDS store is a contiguous 2x b128.
        {
            const int n   = tid & 127;
            const int kb2 = (tid >> 7) * 16;   // 0 or 16
            const float* src = W + (size_t)(ks + kb2) * NPROJ + nBase + n;
            float f[16];
#pragma unroll
            for (int j = 0; j < 16; ++j) f[j] = src[(size_t)j * NPROJ];
            uint4 u0, u1;
            u0.x = pk2(f[0], f[1]);   u0.y = pk2(f[2], f[3]);
            u0.z = pk2(f[4], f[5]);   u0.w = pk2(f[6], f[7]);
            u1.x = pk2(f[8], f[9]);   u1.y = pk2(f[10], f[11]);
            u1.z = pk2(f[12], f[13]); u1.w = pk2(f[14], f[15]);
            uint4* dst = (uint4*)(Ws + n * 32 + kb2);
            dst[0] = u0; dst[1] = u1;
        }
        // Prefetch next K tile into cache while we compute this one.
        if (ks + 32 < DIM) {
            __builtin_prefetch(x + (size_t)(mBase + (tid >> 1)) * DIM + ks + 32 + (tid & 1) * 16, 0, 1);
            __builtin_prefetch(W + (size_t)(ks + 32 + (tid >> 3)) * NPROJ + nBase + (tid & 7) * 16, 0, 1);
        }
        __syncthreads();

        const v16h a0 = frag16(Xs + (wm +      (lane & 15)) * 32, lane);
        const v16h a1 = frag16(Xs + (wm + 16 + (lane & 15)) * 32, lane);
#pragma unroll
        for (int st = 0; st < 4; ++st) {
            const v16h bf = frag16(Ws + (wn + st * 16 + (lane & 15)) * 32, lane);
            acc[0][st] = __builtin_amdgcn_wmma_f32_16x16x32_f16(
                false, a0, false, bf, (short)0, acc[0][st], false, false);
            acc[1][st] = __builtin_amdgcn_wmma_f32_16x16x32_f16(
                false, a1, false, bf, (short)0, acc[1][st], false, false);
        }
        __syncthreads();
    }

    // Epilogue: bias + RoPE (table lookup) + pack f16 q/k.
    const int laneN  = lane & 15;
    const int laneHi = lane >> 4;
#pragma unroll
    for (int st = 0; st < 4; ++st) {
        const int c    = nBase + wn + st * 16 + laneN;  // global proj column
        const int h    = c >> 7;                        // head
        const int t    = c & 127;
        const int sidx = t & 63;                        // index within head vec
        const bool isk = (t >= 64);
        const int  ti  = sidx >> 1;                     // rope table column
        const float bv = bias[c];
#pragma unroll
        for (int mi = 0; mi < 2; ++mi) {
#pragma unroll
            for (int r = 0; r < 8; ++r) {
                const int m    = mBase + wm + mi * 16 + r + laneHi * 8;
                const int bidx = m >> 9;                // / 512
                const int l    = m & 511;               // seq position
                float v  = acc[mi][st][r] + bv;
                float pv = __shfl_xor(v, 1, 32);        // paired feature (col ^ 1)
                const float2 sc = tab[l * 32 + ti];     // (sin, cos)
                const float o = (sidx & 1) ? (v * sc.y + pv * sc.x)
                                           : (v * sc.y - pv * sc.x);
                _Float16* dst = (isk ? kb : qb) +
                    ((((size_t)bidx * HEADS + h) * LSEQ + l) * HSZ + sidx);
                *dst = (_Float16)o;
            }
        }
    }
}

// ---------------------------------------------------------------------------
// Kernel 2: logits[b,h,m,n] = (mask/causal(Q·K^T)) / 8.
// Block = 256 threads (8 waves). Output tile 128x128 per (b,h).
// ---------------------------------------------------------------------------
__global__ __launch_bounds__(256) void attn_logits_kernel(
    const _Float16* __restrict__ qb, const _Float16* __restrict__ kb,
    const float* __restrict__ mask, float* __restrict__ out) {
    __shared__ __align__(16) _Float16 Qs[128 * 64];
    __shared__ __align__(16) _Float16 Ks[128 * 64];

    const int tid  = threadIdx.x;
    const int lane = tid & 31;
    const int w    = tid >> 5;
    const int bh   = blockIdx.z;
    const int bidx = bh / HEADS;
    const int mBase = blockIdx.y * 128;
    const int nBase = blockIdx.x * 128;

    // Stage Q and K tiles (128 x 64 halves = 16 KB each), b128 vector copies.
    {
        const uint4* qsrc = (const uint4*)(qb + ((size_t)bh * LSEQ + mBase) * HSZ);
        const uint4* ksrc = (const uint4*)(kb + ((size_t)bh * LSEQ + nBase) * HSZ);
        uint4* qd = (uint4*)Qs;
        uint4* kd = (uint4*)Ks;
#pragma unroll
        for (int i = 0; i < 4; ++i) {
            const int idx = tid + i * 256;   // 1024 uint4 per tile
            qd[idx] = qsrc[idx];
            kd[idx] = ksrc[idx];
        }
    }
    __syncthreads();

    const int wm = (w & 3) * 32;    // wave: 32 rows
    const int wn = (w >> 2) * 64;   // wave: 64 cols
    v8f acc[2][4] = {};

#pragma unroll
    for (int kk = 0; kk < 2; ++kk) {   // K = 64 -> two 32-wide steps
        const v16h a0 = frag16(Qs + (wm +      (lane & 15)) * 64 + kk * 32, lane);
        const v16h a1 = frag16(Qs + (wm + 16 + (lane & 15)) * 64 + kk * 32, lane);
#pragma unroll
        for (int ni = 0; ni < 4; ++ni) {
            const v16h bf = frag16(Ks + (wn + ni * 16 + (lane & 15)) * 64 + kk * 32, lane);
            acc[0][ni] = __builtin_amdgcn_wmma_f32_16x16x32_f16(
                false, a0, false, bf, (short)0, acc[0][ni], false, false);
            acc[1][ni] = __builtin_amdgcn_wmma_f32_16x16x32_f16(
                false, a1, false, bf, (short)0, acc[1][ni], false, false);
        }
    }

    // Epilogue: mask, strict-lower-tri -NEG, scale 1/8, store fp32.
    const int laneN  = lane & 15;
    const int laneHi = lane >> 4;
#pragma unroll
    for (int mi = 0; mi < 2; ++mi) {
#pragma unroll
        for (int ni = 0; ni < 4; ++ni) {
            const int col = nBase + wn + ni * 16 + laneN;
            const float m3 = mask[bidx * LSEQ + col];
#pragma unroll
            for (int r = 0; r < 8; ++r) {
                const int row = mBase + wm + mi * 16 + r + laneHi * 8;
                const float m2 = mask[bidx * LSEQ + row];
                float v = acc[mi][ni][r];
                v = v * m2 + (1.0f - m2) * (-NEGC);
                v = v * m3 + (1.0f - m3) * (-NEGC);
                if (row > col) v -= NEGC;
                out[((size_t)bh * LSEQ + row) * LSEQ + col] = v * 0.125f;
            }
        }
    }
}

extern "C" void kernel_launch(void* const* d_in, const int* in_sizes, int n_in,
                              void* d_out, int out_size, void* d_ws, size_t ws_size,
                              hipStream_t stream) {
    const float* x    = (const float*)d_in[0];   // (32, 512, 768)
    const float* mask = (const float*)d_in[1];   // (32, 512)
    const float* W    = (const float*)d_in[2];   // (768, 1536)
    const float* bias = (const float*)d_in[3];   // (1536,)
    float* out = (float*)d_out;                  // (32, 12, 512*512)

    const size_t qkElems = (size_t)BATCH * HEADS * LSEQ * HSZ;  // 12.58M halves
    _Float16* qb = (_Float16*)d_ws;
    _Float16* kb = qb + qkElems;
    float2*  tab = (float2*)(kb + qkElems);      // 512*32 float2 = 128 KB

    rope_table_kernel<<<dim3(LSEQ), dim3(32), 0, stream>>>(tab);

    dim3 g1((BATCH * LSEQ) / 128, NPROJ / 128);  // 128 x 12 blocks
    proj_rope_kernel<<<g1, 256, 0, stream>>>(x, W, bias, tab, qb, kb);

    dim3 g2(LSEQ / 128, LSEQ / 128, BATCH * HEADS);  // 4 x 4 x 384 blocks
    attn_logits_kernel<<<g2, 256, 0, stream>>>(qb, kb, mask, out);
}
